// Decoder_6579889898147
// MI455X (gfx1250) — matrix operand order repaired and checked by reading, not compiled, not run on hardware
//
#include <hip/hip_runtime.h>
#include <math.h>

// ---------------------------------------------------------------------------
// LSTM caption decoder for MI455X (gfx1250), wave32 + WMMA bf16 path.
// f32 operands pre-split ONCE into bf16 (hi, lo); GEMM inner loops consume
// bf16 via b128 loads and compute acc += Ah*Bh + Ah*Bl + Al*Bh with
// v_wmma_f32_16x16x32_bf16 (~fp32 accuracy at bf16-WMMA throughput).
// Throughput GEMMs use 2x2 register blocking (32x32 per wave) for 2x
// fragment reuse; the latency-bound recurrent step keeps short WMMA chains.
// ---------------------------------------------------------------------------

typedef __attribute__((ext_vector_type(16))) __bf16 v16bf;
typedef __attribute__((ext_vector_type(8)))  __bf16 v8bf;
typedef __attribute__((ext_vector_type(8)))  float  v8f;

__device__ __forceinline__ unsigned short f2bf(float f) {
  unsigned int u = __float_as_uint(f);
  u += 0x7fffu + ((u >> 16) & 1u);      // round-to-nearest-even
  return (unsigned short)(u >> 16);
}
__device__ __forceinline__ float bf2f(unsigned short s) {
  return __uint_as_float(((unsigned int)s) << 16);
}
__device__ __forceinline__ void split1(float f, unsigned short& hi,
                                       unsigned short& lo) {
  hi = f2bf(f);
  lo = f2bf(f - bf2f(hi));
}

// A fragment (16x32 bf16, MxK): lane holds row l%16; K runs
// {k0+8*half .. +7} then {k0+16+8*half .. +7}. Two b128 loads.
__device__ __forceinline__ v16bf loadA16(const unsigned short* __restrict__ row,
                                         int k0, int half) {
  union { v8bf h[2]; v16bf v; } u;
  const __bf16* p = (const __bf16*)row;
  u.h[0] = *(const v8bf*)(p + k0 + half * 8);
  u.h[1] = *(const v8bf*)(p + k0 + 16 + half * 8);
  return u.v;
}
// B fragment (32x16 bf16, KxN): lane holds column l%16; K runs
// {k0+8*half .. +7} then {k0+16+8*half .. +7}. Two b128 loads.
__device__ __forceinline__ v16bf loadB16(const unsigned short* __restrict__ row,
                                         int k0, int half) {
  union { v8bf h[2]; v16bf v; } u;
  const __bf16* p = (const __bf16*)row;
  u.h[0] = *(const v8bf*)(p + k0 + half * 8);
  u.h[1] = *(const v8bf*)(p + k0 + 16 + half * 8);
  return u.v;
}

// hi/lo split product: acc += Ah*Bh + Ah*Bl + Al*Bh  (f32 accumulate)
__device__ __forceinline__ v8f wmma3(v16bf ah, v16bf al, v16bf bh, v16bf bl,
                                     v8f acc) {
  acc = __builtin_amdgcn_wmma_f32_16x16x32_bf16(false, ah, false, bh,
                                                (short)0, acc, false, false);
  acc = __builtin_amdgcn_wmma_f32_16x16x32_bf16(false, ah, false, bl,
                                                (short)0, acc, false, false);
  acc = __builtin_amdgcn_wmma_f32_16x16x32_bf16(false, al, false, bh,
                                                (short)0, acc, false, false);
  return acc;
}

__device__ __forceinline__ float sigmoidf_(float x) {
  return 1.0f / (1.0f + __expf(-x));
}

// Problem constants
#define BB  128
#define TT  20
#define TS  19               // T-1 unrolled steps
#define EE  512
#define HH  512
#define VV  10000
#define G4H 2048             // 4*H
#define MROWS (BB * TS)      // 2432
#define MT    (MROWS / 16)   // 152 M-tiles
#define MP    (MT / 2)       // 76 M-tile pairs
#define NT_GX (G4H / 16)     // 128 N-tiles (gates)
#define NP_GX (NT_GX / 2)    // 64 pairs
#define NT_V  (VV / 16)      // 625 N-tiles (vocab)
#define NP_V  ((NT_V + 1) / 2) // 313 pairs (last pair half-valid)

// ---------------------------------------------------------------------------
__global__ void k_zero(float* __restrict__ p, int n) {
  int i = blockIdx.x * blockDim.x + threadIdx.x;
  if (i < n) p[i] = 0.0f;
}

// One-time f32 -> bf16(hi,lo) split of a flat array (weights).
__global__ void k_cvt(const float* __restrict__ src,
                      unsigned short* __restrict__ hi,
                      unsigned short* __restrict__ lo, int n) {
  int i = blockIdx.x * blockDim.x + threadIdx.x;
  if (i < n) split1(src[i], hi[i], lo[i]);
}

// Build xs[m,k] (m = b*19+t): t==0 -> features[b], else emb[captions[b,t]].
__global__ void k_build_xs(const float* __restrict__ features,
                           const int* __restrict__ captions,
                           const float* __restrict__ emb,
                           unsigned short* __restrict__ xs_hi,
                           unsigned short* __restrict__ xs_lo) {
  int i = blockIdx.x * blockDim.x + threadIdx.x;   // over 2432*512
  if (i >= MROWS * EE) return;
  int m = i / EE, k = i - m * EE;
  int t = m % TS, b = m / TS;
  float v = (t == 0) ? features[(size_t)b * EE + k]
                     : emb[(size_t)captions[b * TT + t] * EE + k];
  split1(v, xs_hi[i], xs_lo[i]);
}

// ---------------------------------------------------------------------------
// gx[m,:] = xs[m] @ W_ih^T + b_ih + b_hh   (M=2432, N=2048, K=512)
// 2x2 register blocking: each wave computes a 32x32 block (4 accumulators).
__global__ void k_embed_gx(const unsigned short* __restrict__ xs_hi,
                           const unsigned short* __restrict__ xs_lo,
                           const unsigned short* __restrict__ w_hi,
                           const unsigned short* __restrict__ w_lo,
                           const float* __restrict__ b_ih,
                           const float* __restrict__ b_hh,
                           float* __restrict__ gx) {
  const int lane = threadIdx.x & 31;
  const int wave = threadIdx.x >> 5;
  const int tile = blockIdx.x * 8 + wave;      // 0 .. 76*64-1
  const int pm = tile % MP;                    // 0..75
  const int pn = tile / MP;                    // 0..63
  const int half = lane >> 4;
  const int lrow = lane & 15;

  const int m0 = pm * 32 + lrow;               // A rows (two tiles)
  const int n0 = pn * 32 + lrow;               // B cols (two tiles)
  const unsigned short* a0h = xs_hi + (size_t)m0 * EE;
  const unsigned short* a0l = xs_lo + (size_t)m0 * EE;
  const unsigned short* a1h = a0h + (size_t)16 * EE;
  const unsigned short* a1l = a0l + (size_t)16 * EE;
  const unsigned short* b0h = w_hi + (size_t)n0 * EE;
  const unsigned short* b0l = w_lo + (size_t)n0 * EE;
  const unsigned short* b1h = b0h + (size_t)16 * EE;
  const unsigned short* b1l = b0l + (size_t)16 * EE;

  v8f c00 = {}, c01 = {}, c10 = {}, c11 = {};
  for (int k0 = 0; k0 < EE; k0 += 32) {
    v16bf A0h = loadA16(a0h, k0, half), A0l = loadA16(a0l, k0, half);
    v16bf A1h = loadA16(a1h, k0, half), A1l = loadA16(a1l, k0, half);
    v16bf B0h = loadB16(b0h, k0, half), B0l = loadB16(b0l, k0, half);
    v16bf B1h = loadB16(b1h, k0, half), B1l = loadB16(b1l, k0, half);
    c00 = wmma3(A0h, A0l, B0h, B0l, c00);
    c01 = wmma3(A0h, A0l, B1h, B1l, c01);
    c10 = wmma3(A1h, A1l, B0h, B0l, c10);
    c11 = wmma3(A1h, A1l, B1h, B1l, c11);
  }

  const int nA = pn * 32 + lrow;               // C: lane owns column
  const int nB = nA + 16;
  const float biasA = b_ih[nA] + b_hh[nA];
  const float biasB = b_ih[nB] + b_hh[nB];
#pragma unroll
  for (int r = 0; r < 8; ++r) {
    const int mr0 = pm * 32 + half * 8 + r;
    const int mr1 = mr0 + 16;
    gx[(size_t)mr0 * G4H + nA] = c00[r] + biasA;
    gx[(size_t)mr0 * G4H + nB] = c01[r] + biasB;
    gx[(size_t)mr1 * G4H + nA] = c10[r] + biasA;
    gx[(size_t)mr1 * G4H + nB] = c11[r] + biasB;
  }
}

// ---------------------------------------------------------------------------
// One recurrent step: gates = gx[:,t,:] + h_{t-1} @ W_hh^T ; LSTM cell.
// h_{t-1} read from hs rows (b*19+t-1) in bf16 hi/lo; new h written to rows
// (b*19+t). t==0 skips the GEMM (h0 == 0). A fragment reused across 4 gates.
__global__ void k_lstm_step(const unsigned short* __restrict__ whh_hi,
                            const unsigned short* __restrict__ whh_lo,
                            const float* __restrict__ gx,
                            unsigned short* __restrict__ hs_hi,
                            unsigned short* __restrict__ hs_lo,
                            float* __restrict__ cstate,
                            int t) {
  const int lane = threadIdx.x & 31;
  const int wave = threadIdx.x >> 5;
  const int idx = blockIdx.x * 8 + wave;   // 0..255
  const int bTile = idx & 7;               // 8 batch tiles
  const int jTile = idx >> 3;              // 32 hidden tiles
  const int half = lane >> 4;
  const int lrow = lane & 15;
  const int j0 = jTile * 16;

  v8f ai = {}, af = {}, ag = {}, ao = {};
  if (t > 0) {
    const size_t arow = ((size_t)(bTile * 16 + lrow) * TS + (t - 1)) * HH;
    const unsigned short* ah_p = hs_hi + arow;
    const unsigned short* al_p = hs_lo + arow;
    const unsigned short* bi_h = whh_hi + (size_t)(0 * HH + j0 + lrow) * HH;
    const unsigned short* bi_l = whh_lo + (size_t)(0 * HH + j0 + lrow) * HH;
    const unsigned short* bf_h = whh_hi + (size_t)(1 * HH + j0 + lrow) * HH;
    const unsigned short* bf_l = whh_lo + (size_t)(1 * HH + j0 + lrow) * HH;
    const unsigned short* bg_h = whh_hi + (size_t)(2 * HH + j0 + lrow) * HH;
    const unsigned short* bg_l = whh_lo + (size_t)(2 * HH + j0 + lrow) * HH;
    const unsigned short* bo_h = whh_hi + (size_t)(3 * HH + j0 + lrow) * HH;
    const unsigned short* bo_l = whh_lo + (size_t)(3 * HH + j0 + lrow) * HH;
    for (int k0 = 0; k0 < HH; k0 += 32) {
      v16bf Ah = loadA16(ah_p, k0, half);
      v16bf Al = loadA16(al_p, k0, half);
      ai = wmma3(Ah, Al, loadB16(bi_h, k0, half), loadB16(bi_l, k0, half), ai);
      af = wmma3(Ah, Al, loadB16(bf_h, k0, half), loadB16(bf_l, k0, half), af);
      ag = wmma3(Ah, Al, loadB16(bg_h, k0, half), loadB16(bg_l, k0, half), ag);
      ao = wmma3(Ah, Al, loadB16(bo_h, k0, half), loadB16(bo_l, k0, half), ao);
    }
  }

  const int j = j0 + lrow;                 // C: lane owns column j
#pragma unroll
  for (int r = 0; r < 8; ++r) {
    const int b = bTile * 16 + half * 8 + r;
    const size_t grow = ((size_t)b * TS + t) * G4H;
    const float gi = ai[r] + gx[grow + 0 * HH + j];
    const float gf = af[r] + gx[grow + 1 * HH + j];
    const float gg = ag[r] + gx[grow + 2 * HH + j];
    const float go = ao[r] + gx[grow + 3 * HH + j];
    const float si = sigmoidf_(gi);
    const float sf = sigmoidf_(gf);
    const float so = sigmoidf_(go);
    const float tg = tanhf(gg);
    const size_t cidx = (size_t)b * HH + j;
    const float c = sf * cstate[cidx] + si * tg;
    cstate[cidx] = c;
    const float h = so * tanhf(c);
    const size_t hidx = ((size_t)b * TS + t) * HH + j;
    unsigned short hh, hl;
    split1(h, hh, hl);
    hs_hi[hidx] = hh;
    hs_lo[hidx] = hl;
  }
}

// ---------------------------------------------------------------------------
// out[m,v] = hs[m] @ W_out^T + b_out   (M=2432, N=10000=625*16, K=512)
// 2x2 register blocking; last N pair has only one valid tile (wave-uniform
// guard so EXEC stays all-ones for WMMA).
__global__ void k_out_proj(const unsigned short* __restrict__ hs_hi,
                           const unsigned short* __restrict__ hs_lo,
                           const unsigned short* __restrict__ w_hi,
                           const unsigned short* __restrict__ w_lo,
                           const float* __restrict__ b_out,
                           float* __restrict__ out) {
  const int lane = threadIdx.x & 31;
  const int wave = threadIdx.x >> 5;
  const int tile = blockIdx.x * 8 + wave;
  if (tile >= MP * NP_V) return;               // grid round-up tail
  const int pm = tile % MP;                    // 0..75
  const int pn = tile / MP;                    // 0..312
  const int half = lane >> 4;
  const int lrow = lane & 15;
  const bool n1ok = (pn * 2 + 1) < NT_V;       // wave-uniform

  const int m0 = pm * 32 + lrow;
  const int n0 = pn * 32 + lrow;
  const unsigned short* a0h = hs_hi + (size_t)m0 * HH;
  const unsigned short* a0l = hs_lo + (size_t)m0 * HH;
  const unsigned short* a1h = a0h + (size_t)16 * HH;
  const unsigned short* a1l = a0l + (size_t)16 * HH;
  const unsigned short* b0h = w_hi + (size_t)n0 * HH;
  const unsigned short* b0l = w_lo + (size_t)n0 * HH;
  const unsigned short* b1h = n1ok ? b0h + (size_t)16 * HH : b0h;
  const unsigned short* b1l = n1ok ? b0l + (size_t)16 * HH : b0l;

  v8f c00 = {}, c01 = {}, c10 = {}, c11 = {};
  for (int k0 = 0; k0 < HH; k0 += 32) {
    v16bf A0h = loadA16(a0h, k0, half), A0l = loadA16(a0l, k0, half);
    v16bf A1h = loadA16(a1h, k0, half), A1l = loadA16(a1l, k0, half);
    v16bf B0h = loadB16(b0h, k0, half), B0l = loadB16(b0l, k0, half);
    v16bf B1h = loadB16(b1h, k0, half), B1l = loadB16(b1l, k0, half);
    c00 = wmma3(A0h, A0l, B0h, B0l, c00);
    c01 = wmma3(A0h, A0l, B1h, B1l, c01);
    c10 = wmma3(A1h, A1l, B0h, B0l, c10);
    c11 = wmma3(A1h, A1l, B1h, B1l, c11);
  }

  const int nA = pn * 32 + lrow;
  const int nB = n1ok ? nA + 16 : nA;
  const float biasA = b_out[nA];
  const float biasB = b_out[nB];
#pragma unroll
  for (int r = 0; r < 8; ++r) {
    const int mr0 = pm * 32 + half * 8 + r;
    const int mr1 = mr0 + 16;
    out[(size_t)mr0 * VV + nA] = c00[r] + biasA;
    out[(size_t)mr1 * VV + nA] = c10[r] + biasA;
    if (n1ok) {
      out[(size_t)mr0 * VV + nB] = c01[r] + biasB;
      out[(size_t)mr1 * VV + nB] = c11[r] + biasB;
    }
  }
}

// ---------------------------------------------------------------------------
extern "C" void kernel_launch(void* const* d_in, const int* in_sizes, int n_in,
                              void* d_out, int out_size, void* d_ws,
                              size_t ws_size, hipStream_t stream) {
  const float* features = (const float*)d_in[0];
  const int*   captions = (const int*)d_in[1];
  const float* emb      = (const float*)d_in[2];
  const float* W_ih     = (const float*)d_in[3];
  const float* W_hh     = (const float*)d_in[4];
  const float* b_ih     = (const float*)d_in[5];
  const float* b_hh     = (const float*)d_in[6];
  const float* W_out    = (const float*)d_in[7];
  const float* b_out    = (const float*)d_in[8];
  float* out = (float*)d_out;

  // Workspace layout: f32 region first, then bf16 (ushort) hi/lo buffers.
  const size_t N_WIH  = (size_t)G4H * EE;     // 1,048,576
  const size_t N_WHH  = (size_t)G4H * HH;     // 1,048,576
  const size_t N_WOUT = (size_t)VV * HH;      // 5,120,000
  const size_t N_XS   = (size_t)MROWS * EE;   // 1,245,184
  const size_t N_HS   = (size_t)MROWS * HH;   // 1,245,184

  float* gx = (float*)d_ws;                          // 2432*2048 f32
  float* c  = gx + (size_t)MROWS * G4H;              // 128*512 f32
  unsigned short* u = (unsigned short*)(c + (size_t)BB * HH);
  unsigned short* wih_hi  = u;              u += N_WIH;
  unsigned short* wih_lo  = u;              u += N_WIH;
  unsigned short* whh_hi  = u;              u += N_WHH;
  unsigned short* whh_lo  = u;              u += N_WHH;
  unsigned short* wout_hi = u;              u += N_WOUT;
  unsigned short* wout_lo = u;              u += N_WOUT;
  unsigned short* xs_hi   = u;              u += N_XS;
  unsigned short* xs_lo   = u;              u += N_XS;
  unsigned short* hs_hi   = u;              u += N_HS;
  unsigned short* hs_lo   = u;              u += N_HS;

  // Zero cell state.
  k_zero<<<(BB * HH + 255) / 256, 256, 0, stream>>>(c, BB * HH);

  // One-time operand conversions (bandwidth-bound, each element touched once).
  k_cvt<<<(int)((N_WIH + 255) / 256), 256, 0, stream>>>(W_ih, wih_hi, wih_lo,
                                                        (int)N_WIH);
  k_cvt<<<(int)((N_WHH + 255) / 256), 256, 0, stream>>>(W_hh, whh_hi, whh_lo,
                                                        (int)N_WHH);
  k_cvt<<<(int)((N_WOUT + 255) / 256), 256, 0, stream>>>(W_out, wout_hi,
                                                         wout_lo, (int)N_WOUT);
  k_build_xs<<<(int)((N_XS + 255) / 256), 256, 0, stream>>>(features, captions,
                                                            emb, xs_hi, xs_lo);

  // Input-side GEMM + biases: 76 x 64 pairs (32x32 per wave), 8 waves/block.
  k_embed_gx<<<(MP * NP_GX) / 8, 256, 0, stream>>>(
      xs_hi, xs_lo, wih_hi, wih_lo, b_ih, b_hh, gx);

  // 19 serialized recurrent steps; h chained through hs in bf16 hi/lo form.
  for (int t = 0; t < TS; ++t) {
    k_lstm_step<<<32, 256, 0, stream>>>(whh_hi, whh_lo, gx, hs_hi, hs_lo, c, t);
  }

  // Vocab projection: 76 x 313 pairs (32x32 per wave).
  k_out_proj<<<(MP * NP_V + 7) / 8, 256, 0, stream>>>(hs_hi, hs_lo, wout_hi,
                                                      wout_lo, b_out, out);
}
